// HymbaBlock_40561671144014
// MI455X (gfx1250) — matrix operand order repaired and checked by reading, not compiled, not run on hardware
//
#include <hip/hip_runtime.h>
#include <hip/hip_bf16.h>
#include <math.h>

typedef __attribute__((ext_vector_type(16))) __bf16 v16bf;
typedef __attribute__((ext_vector_type(8)))  float  v8f;

#define B_    2
#define L_    1024
#define D_    2048
#define I_    4096
#define H_    32
#define HD_   128
#define DS_   16
#define DC_   4
#define DTR_  128
#define XDBC_ 160
#define BL_   (B_ * L_)
#define EPS_  1e-6f

// ---------------------------------------------------------------------------
// CDNA5 async global->LDS copy (ASYNCcnt path, bypasses VGPRs)
// ---------------------------------------------------------------------------
__device__ __forceinline__ void async_b128(unsigned lds_addr, const void* gptr) {
  asm volatile("global_load_async_to_lds_b128 %0, %1, off"
               :: "v"(lds_addr), "v"(gptr)
               : "memory");
}

// ---------------------------------------------------------------------------
// fp32 -> bf16 conversion
// ---------------------------------------------------------------------------
__global__ __launch_bounds__(256) void cvt_f32_bf16(const float* __restrict__ src,
                                                    __bf16* __restrict__ dst,
                                                    size_t n) {
  size_t base = (size_t)blockIdx.x * 1024 + threadIdx.x;
#pragma unroll
  for (int j = 0; j < 4; ++j) {
    size_t i = base + (size_t)j * 256;
    if (i < n) dst[i] = (__bf16)src[i];
  }
}

// ---------------------------------------------------------------------------
// Tiled bf16 WMMA GEMM:  C[M,N] (f32) = A[M,K] (bf16, lda) * W[N,K]^T (bf16, ldw)
// Block tile 128x128, K-step 32, double-buffered LDS filled by
// GLOBAL_LOAD_ASYNC_TO_LDS_B128 so the next K-slice DMA overlaps WMMA work.
// 8 waves: 2(M) x 4(N), each wave 64x32 = 8 WMMAs per K-step.
// Requires: M multiple of 128, K multiple of 32, lda/ldw multiples of 8.
// ---------------------------------------------------------------------------
__global__ __launch_bounds__(256) void gemm_bf16_wmma(
    const __bf16* __restrict__ A, int lda,
    const __bf16* __restrict__ W, int ldw,
    float* __restrict__ C, int ldc,
    int M, int N, int K) {
  __shared__ __bf16 As[2][128][48];  // 48 = 32 + 16 pad (keeps 16B alignment)
  __shared__ __bf16 Bs[2][128][48];

  const int tid  = threadIdx.x;
  const int lane = tid & 31;
  const int wave = tid >> 5;
  const int half = lane >> 4;   // 0 | 1
  const int ml   = lane & 15;

  const int bm = blockIdx.y * 128;
  const int bn = blockIdx.x * 128;
  const int wm = (wave >> 2) * 64;  // 0 | 64
  const int wn = (wave & 3) * 32;   // 0..96

  v8f acc[4][2] = {};

  const int lr = tid >> 1;        // 0..127 tile row handled by this thread
  const int ls = (tid & 1) * 16;  // 0 | 16 (column segment)

  // clamped B source row (cols >= N are never stored, so duplicated data is fine)
  int nrow = bn + lr;
  if (nrow > N - 1) nrow = N - 1;

  // per-thread async staging of one (A,B) K-slice into LDS buffer `buf`
  auto issue_tile = [&](int buf, int kk) {
    const __bf16* ga = A + (size_t)(bm + lr) * lda + kk + ls;
    unsigned la = (unsigned)(unsigned long long)&As[buf][lr][ls];
    async_b128(la, ga);
    async_b128(la + 16u, ga + 8);
    const __bf16* gb = W + (size_t)nrow * ldw + kk + ls;
    unsigned lb = (unsigned)(unsigned long long)&Bs[buf][lr][ls];
    async_b128(lb, gb);
    async_b128(lb + 16u, gb + 8);
  };

  const int nIter = K >> 5;
  issue_tile(0, 0);

  for (int it = 0; it < nIter; ++it) {
    const int cur = it & 1;
    if (it + 1 < nIter) {
      issue_tile(cur ^ 1, (it + 1) << 5);
      // 4 next-slice ops in flight; per-wave in-order completion =>
      // ASYNCcnt<=4 means the current slice's 4 ops have landed in LDS.
      asm volatile("s_wait_asynccnt 0x4" ::: "memory");
    } else {
      asm volatile("s_wait_asynccnt 0x0" ::: "memory");
    }
    __syncthreads();

    // ---- A fragments (ISA 16-bit A 16x32 layout) ----
    v16bf afr[4];
#pragma unroll
    for (int mt = 0; mt < 4; ++mt) {
      const __bf16* p = &As[cur][wm + mt * 16 + ml][0];
#pragma unroll
      for (int v = 0; v < 8; ++v) {
        int k = ((v >> 2) << 4) + (half << 3) + ((v & 3) << 1);
        afr[mt][2 * v]     = p[k];
        afr[mt][2 * v + 1] = p[k + 1];
      }
    }
    // ---- B fragments (32x16, lane = N, halves split K) ----
    v16bf bfr[2];
#pragma unroll
    for (int bt = 0; bt < 2; ++bt) {
      const __bf16* p = &Bs[cur][wn + bt * 16 + ml][0];
#pragma unroll
      for (int v = 0; v < 8; ++v) {
        int k = ((v >> 2) << 4) + (half << 3) + ((v & 3) << 1);
        bfr[bt][2 * v]     = p[k];
        bfr[bt][2 * v + 1] = p[k + 1];
      }
    }
#pragma unroll
    for (int mt = 0; mt < 4; ++mt)
#pragma unroll
      for (int bt = 0; bt < 2; ++bt)
        acc[mt][bt] = __builtin_amdgcn_wmma_f32_16x16x32_bf16(
            false, afr[mt], false, bfr[bt], (short)0, acc[mt][bt], false, false);
    __syncthreads();  // protects LDS buffer `cur` from next iteration's DMA
  }

  // ---- store C (VGPR r -> row r / r+8, lane -> col) ----
#pragma unroll
  for (int mt = 0; mt < 4; ++mt)
#pragma unroll
    for (int bt = 0; bt < 2; ++bt)
#pragma unroll
      for (int r = 0; r < 8; ++r) {
        int row = bm + wm + mt * 16 + r + half * 8;
        int col = bn + wn + bt * 16 + ml;
        if (col < N) C[(size_t)row * ldc + col] = acc[mt][bt][r];
      }
}

// ---------------------------------------------------------------------------
// Depthwise causal conv (DC=4) + SiLU over x half of proj; emits f32 + bf16
// ---------------------------------------------------------------------------
__global__ __launch_bounds__(256) void conv_silu_kernel(
    const float* __restrict__ proj, const float* __restrict__ cw,
    const float* __restrict__ cb, float* __restrict__ xc,
    __bf16* __restrict__ xcbf) {
  size_t idx = (size_t)blockIdx.x * 256 + threadIdx.x;  // over BL*I
  if (idx >= (size_t)BL_ * I_) return;
  int i  = (int)(idx % I_);
  int bl = (int)(idx / I_);
  int l  = bl % L_;
  int b0 = bl - l;  // b*L
  float acc = cb[i];
#pragma unroll
  for (int j = 0; j < DC_; ++j) {
    int ls = l - (DC_ - 1) + j;
    if (ls >= 0)
      acc += proj[(size_t)(b0 + ls) * (2 * I_) + i] * cw[i * DC_ + j];
  }
  float s = acc / (1.f + __expf(-acc));
  xc[idx]   = s;
  xcbf[idx] = (__bf16)s;
}

// ---------------------------------------------------------------------------
// dt = softplus(gemm_out + bias)   (in place on gemm output)
// ---------------------------------------------------------------------------
__global__ __launch_bounds__(256) void dt_softplus_kernel(
    float* __restrict__ dt, const float* __restrict__ dtb) {
  size_t idx = (size_t)blockIdx.x * 256 + threadIdx.x;
  if (idx >= (size_t)BL_ * I_) return;
  float x = dt[idx] + dtb[idx % I_];
  dt[idx] = (x > 20.f) ? x : log1pf(__expf(x));
}

// ---------------------------------------------------------------------------
// Selective-scan: one thread per (b, i) channel, 16-state in registers.
// global_prefetch for the next timestep's rows hides latency.
// ---------------------------------------------------------------------------
__global__ __launch_bounds__(256) void scan_kernel(
    const float* __restrict__ dt, const float* __restrict__ xc,
    const float* __restrict__ xdbc, const float* __restrict__ A_log,
    const float* __restrict__ Dsk, float* __restrict__ ym) {
  int i = blockIdx.x * 256 + threadIdx.x;
  int b = blockIdx.y;
  float Arow[DS_], h[DS_];
#pragma unroll
  for (int s = 0; s < DS_; ++s) {
    Arow[s] = -__expf(A_log[i * DS_ + s]);
    h[s] = 0.f;
  }
  float dsk = Dsk[i];
  for (int l = 0; l < L_; ++l) {
    size_t bl = (size_t)b * L_ + l;
    if (l + 1 < L_) {
      __builtin_prefetch(&dt[(bl + 1) * I_ + i], 0, 1);
      __builtin_prefetch(&xc[(bl + 1) * I_ + i], 0, 1);
      __builtin_prefetch(&xdbc[(bl + 1) * XDBC_ + DTR_], 0, 1);
    }
    float dt_t = dt[bl * I_ + i];
    float x_t  = xc[bl * I_ + i];
    float dbx  = dt_t * x_t;
    const float* xb = xdbc + bl * XDBC_;
    float y = 0.f;
#pragma unroll
    for (int s = 0; s < DS_; ++s) {
      h[s] = h[s] * __expf(dt_t * Arow[s]) + dbx * xb[DTR_ + s];
      y += h[s] * xb[DTR_ + DS_ + s];
    }
    ym[bl * I_ + i] = y + dsk * x_t;
  }
}

// ---------------------------------------------------------------------------
// RoPE on q (=xc heads) and k; pack q/k/v to [B,H,L,HD] bf16
// ---------------------------------------------------------------------------
__global__ __launch_bounds__(256) void rope_pack_kernel(
    const float* __restrict__ xc, const float* __restrict__ kf,
    const float* __restrict__ vf, __bf16* __restrict__ qh,
    __bf16* __restrict__ kh, __bf16* __restrict__ vh) {
  size_t idx = (size_t)blockIdx.x * 256 + threadIdx.x;  // over B*H*L*HD
  if (idx >= (size_t)B_ * H_ * L_ * HD_) return;
  int d    = (int)(idx & (HD_ - 1));
  size_t t = idx >> 7;           // (b*H + h)*L + l
  int l    = (int)(t % L_);
  size_t bh = t / L_;
  int h = (int)(bh % H_);
  int b = (int)(bh / H_);
  size_t src = ((size_t)b * L_ + l) * I_ + (size_t)h * HD_;
  int dj = d & 63;
  float inv = __powf(10000.f, -(float)(2 * dj) / 128.f);
  float ang = (float)l * inv;
  float c = __cosf(ang), s = __sinf(ang);
  float q1 = xc[src + d];
  float qr = (d < 64) ? -xc[src + d + 64] : xc[src + d - 64];
  float k1 = kf[src + d];
  float kr = (d < 64) ? -kf[src + d + 64] : kf[src + d - 64];
  qh[idx] = (__bf16)(q1 * c + qr * s);
  kh[idx] = (__bf16)(k1 * c + kr * s);
  vh[idx] = (__bf16)vf[src + d];
}

// ---------------------------------------------------------------------------
// Causal flash attention with WMMA.  Block = 4 waves, each wave = 16 q-rows.
// 32 keys per iteration (two 16x16 S tiles; P*V is one full 16x16x32 WMMA).
// Writes y_attn directly in (b, l, h*HD+d) layout.
// ---------------------------------------------------------------------------
__global__ __launch_bounds__(128) void flash_attn_kernel(
    const __bf16* __restrict__ Q, const __bf16* __restrict__ Kh,
    const __bf16* __restrict__ Vh, float* __restrict__ Ya) {
  const int bh   = blockIdx.x;      // b*H + h
  const int b    = bh / H_;
  const int h    = bh % H_;
  const int wave = threadIdx.x >> 5;
  const int lane = threadIdx.x & 31;
  const int half = lane >> 4;
  const int ml   = lane & 15;
  const int qbase = blockIdx.y * 64 + wave * 16;

  __shared__ __bf16 Ps[4][16][40];  // per-wave P tile (16x32 + pad)

  const __bf16* Qb = Q  + (size_t)bh * L_ * HD_;
  const __bf16* Kb = Kh + (size_t)bh * L_ * HD_;
  const __bf16* Vb = Vh + (size_t)bh * L_ * HD_;

  // Q A-fragments (16 rows x HD, split over 4 K-chunks of 32)
  v16bf qf[4];
  {
    const __bf16* qp = Qb + (size_t)(qbase + ml) * HD_;
#pragma unroll
    for (int f = 0; f < 4; ++f)
#pragma unroll
      for (int v = 0; v < 8; ++v) {
        int k = f * 32 + ((v >> 2) << 4) + (half << 3) + ((v & 3) << 1);
        qf[f][2 * v]     = qp[k];
        qf[f][2 * v + 1] = qp[k + 1];
      }
  }

  v8f O[8] = {};
  float m_i[8], l_i[8];
#pragma unroll
  for (int r = 0; r < 8; ++r) { m_i[r] = -3.0e38f; l_i[r] = 0.f; }
  const float scale = 0.08838834764831845f;  // 1/sqrt(128)

  const int nkt = (qbase + 16 + 31) >> 5;  // ceil((qbase+16)/32)
  for (int kt = 0; kt < nkt; ++kt) {
    const int kb = kt * 32;
    v8f S0 = {0.f, 0.f, 0.f, 0.f, 0.f, 0.f, 0.f, 0.f};
    v8f S1 = {0.f, 0.f, 0.f, 0.f, 0.f, 0.f, 0.f, 0.f};
#pragma unroll
    for (int f = 0; f < 4; ++f) {
      v16bf b0, b1;
      const __bf16* kp0 = Kb + (size_t)(kb + ml) * HD_;
      const __bf16* kp1 = Kb + (size_t)(kb + 16 + ml) * HD_;
#pragma unroll
      for (int v = 0; v < 8; ++v) {
        int k = f * 32 + ((v >> 2) << 4) + (half << 3) + ((v & 3) << 1);
        b0[2 * v]     = kp0[k];
        b0[2 * v + 1] = kp0[k + 1];
        b1[2 * v]     = kp1[k];
        b1[2 * v + 1] = kp1[k + 1];
      }
      S0 = __builtin_amdgcn_wmma_f32_16x16x32_bf16(false, qf[f], false, b0,
                                                   (short)0, S0, false, false);
      S1 = __builtin_amdgcn_wmma_f32_16x16x32_bf16(false, qf[f], false, b1,
                                                   (short)0, S1, false, false);
    }
    // ---- online softmax over 32 columns ----
#pragma unroll
    for (int r = 0; r < 8; ++r) {
      int qr = qbase + r + half * 8;
      float s0 = S0[r] * scale;
      float s1 = S1[r] * scale;
      if (kb + ml > qr)      s0 = -3.0e38f;
      if (kb + 16 + ml > qr) s1 = -3.0e38f;
      float mx = fmaxf(s0, s1);
#pragma unroll
      for (int d = 1; d < 16; d <<= 1) mx = fmaxf(mx, __shfl_xor(mx, d, 32));
      float mn = fmaxf(m_i[r], mx);
      float p0 = __expf(s0 - mn);
      float p1 = __expf(s1 - mn);
      float rs = p0 + p1;
#pragma unroll
      for (int d = 1; d < 16; d <<= 1) rs += __shfl_xor(rs, d, 32);
      float alpha = __expf(m_i[r] - mn);
      l_i[r] = l_i[r] * alpha + rs;
      m_i[r] = mn;
#pragma unroll
      for (int bt = 0; bt < 8; ++bt) O[bt][r] *= alpha;
      Ps[wave][r + half * 8][ml]      = (__bf16)p0;
      Ps[wave][r + half * 8][16 + ml] = (__bf16)p1;
    }
    // wave-internal LDS RAW: only need a DS-counter wait (no block barrier —
    // waves have divergent trip counts).
    asm volatile("s_wait_dscnt 0x0" ::: "memory");

    // P as A-fragment (16x32)
    v16bf pf;
#pragma unroll
    for (int v = 0; v < 8; ++v) {
      int k = ((v >> 2) << 4) + (half << 3) + ((v & 3) << 1);
      pf[2 * v]     = Ps[wave][ml][k];
      pf[2 * v + 1] = Ps[wave][ml][k + 1];
    }
    // V as B-fragments (32 keys x HD), accumulate O
#pragma unroll
    for (int bt = 0; bt < 8; ++bt) {
      v16bf vfr;
#pragma unroll
      for (int v = 0; v < 8; ++v) {
        int kloc = ((v >> 2) << 4) + (half << 3) + ((v & 3) << 1);
        const __bf16* vp = Vb + (size_t)(kb + kloc) * HD_ + bt * 16 + ml;
        vfr[2 * v]     = vp[0];
        vfr[2 * v + 1] = vp[HD_];
      }
      O[bt] = __builtin_amdgcn_wmma_f32_16x16x32_bf16(false, pf, false, vfr,
                                                      (short)0, O[bt], false, false);
    }
  }

  // ---- epilogue: normalize, write to (b, l, h*HD+d) ----
#pragma unroll
  for (int bt = 0; bt < 8; ++bt)
#pragma unroll
    for (int r = 0; r < 8; ++r) {
      int l   = qbase + r + half * 8;
      int col = h * HD_ + bt * 16 + ml;
      Ya[((size_t)b * L_ + l) * I_ + col] = O[bt][r] / l_i[r];
    }
}

// ---------------------------------------------------------------------------
// Per-head RMSNorm on y_attn and y_mamba, 0.5*(sum), SiLU(z) gate -> bf16
// One wave per (b, l, h); 4 elements per lane.
// ---------------------------------------------------------------------------
__global__ __launch_bounds__(256) void norm_gate_kernel(
    const float* __restrict__ ya, const float* __restrict__ ym,
    const float* __restrict__ aw, const float* __restrict__ mw,
    const float* __restrict__ proj, __bf16* __restrict__ yout) {
  int idx  = blockIdx.x * 8 + (threadIdx.x >> 5);  // over BL*H
  int lane = threadIdx.x & 31;
  int h  = idx % H_;
  int bl = idx / H_;
  size_t base = (size_t)bl * I_ + (size_t)h * HD_;
  float a4[4], m4[4];
  float ssa = 0.f, ssm = 0.f;
#pragma unroll
  for (int j = 0; j < 4; ++j) {
    int d = lane + j * 32;
    a4[j] = ya[base + d];
    m4[j] = ym[base + d];
    ssa += a4[j] * a4[j];
    ssm += m4[j] * m4[j];
  }
#pragma unroll
  for (int d = 1; d < 32; d <<= 1) {
    ssa += __shfl_xor(ssa, d, 32);
    ssm += __shfl_xor(ssm, d, 32);
  }
  float ra = rsqrtf(ssa * (1.f / HD_) + EPS_);
  float rm = rsqrtf(ssm * (1.f / HD_) + EPS_);
#pragma unroll
  for (int j = 0; j < 4; ++j) {
    int d = lane + j * 32;
    float yn = 0.5f * (a4[j] * ra * aw[h * HD_ + d] + m4[j] * rm * mw[h * HD_ + d]);
    float z  = proj[(size_t)bl * (2 * I_) + I_ + h * HD_ + d];
    float g  = z / (1.f + __expf(-z));
    yout[base + d] = (__bf16)(yn * g);
  }
}

// ---------------------------------------------------------------------------
// Host-side orchestration
// ---------------------------------------------------------------------------
extern "C" void kernel_launch(void* const* d_in, const int* in_sizes, int n_in,
                              void* d_out, int out_size, void* d_ws, size_t ws_size,
                              hipStream_t stream) {
  (void)in_sizes; (void)n_in; (void)out_size; (void)ws_size;

  const float* hs       = (const float*)d_in[0];
  const float* in_w     = (const float*)d_in[1];
  const float* conv_w   = (const float*)d_in[2];
  const float* conv_b   = (const float*)d_in[3];
  const float* xproj_w  = (const float*)d_in[4];
  const float* dt_w     = (const float*)d_in[5];
  const float* dt_b     = (const float*)d_in[6];
  const float* A_log    = (const float*)d_in[7];
  const float* Dsk      = (const float*)d_in[8];
  const float* k_w      = (const float*)d_in[9];
  const float* v_w      = (const float*)d_in[10];
  const float* attn_nw  = (const float*)d_in[11];
  const float* mamba_nw = (const float*)d_in[12];
  const float* out_w    = (const float*)d_in[13];
  float* out = (float*)d_out;

  char* base = (char*)d_ws;
  size_t off = 0;
  auto alloc = [&](size_t bytes) -> void* {
    void* p = base + off;
    off += (bytes + 255) & ~(size_t)255;
    return p;
  };

  __bf16* hs_bf   = (__bf16*)alloc((size_t)BL_ * D_ * 2);
  __bf16* inw_bf  = (__bf16*)alloc((size_t)2 * I_ * D_ * 2);
  __bf16* kw_bf   = (__bf16*)alloc((size_t)I_ * D_ * 2);
  __bf16* vw_bf   = (__bf16*)alloc((size_t)I_ * D_ * 2);
  __bf16* xpw_bf  = (__bf16*)alloc((size_t)XDBC_ * I_ * 2);
  __bf16* dtw_bf  = (__bf16*)alloc((size_t)I_ * DTR_ * 2);
  __bf16* ow_bf   = (__bf16*)alloc((size_t)D_ * I_ * 2);
  float*  projf   = (float*)alloc((size_t)BL_ * 2 * I_ * 4);
  float*  xcf     = (float*)alloc((size_t)BL_ * I_ * 4);
  __bf16* xc_bf   = (__bf16*)alloc((size_t)BL_ * I_ * 2);
  float*  xdbcf   = (float*)alloc((size_t)BL_ * XDBC_ * 4);
  __bf16* xdbc_bf = (__bf16*)alloc((size_t)BL_ * XDBC_ * 2);
  float*  dtf     = (float*)alloc((size_t)BL_ * I_ * 4);
  float*  kfb     = (float*)alloc((size_t)BL_ * I_ * 4);
  float*  vfb     = (float*)alloc((size_t)BL_ * I_ * 4);
  float*  ymf     = (float*)alloc((size_t)BL_ * I_ * 4);
  __bf16* qh      = (__bf16*)alloc((size_t)BL_ * I_ * 2);
  __bf16* kh      = (__bf16*)alloc((size_t)BL_ * I_ * 2);
  __bf16* vh      = (__bf16*)alloc((size_t)BL_ * I_ * 2);
  float*  yaf     = (float*)alloc((size_t)BL_ * I_ * 4);
  __bf16* ygate   = (__bf16*)alloc((size_t)BL_ * I_ * 2);

  auto cvt = [&](const float* s, __bf16* d, size_t n) {
    int blocks = (int)((n + 1023) / 1024);
    cvt_f32_bf16<<<blocks, 256, 0, stream>>>(s, d, n);
  };
  cvt(hs, hs_bf, (size_t)BL_ * D_);
  cvt(in_w, inw_bf, (size_t)2 * I_ * D_);
  cvt(k_w, kw_bf, (size_t)I_ * D_);
  cvt(v_w, vw_bf, (size_t)I_ * D_);
  cvt(xproj_w, xpw_bf, (size_t)XDBC_ * I_);
  cvt(dt_w, dtw_bf, (size_t)I_ * DTR_);
  cvt(out_w, ow_bf, (size_t)D_ * I_);

  auto gemm = [&](const __bf16* A, int lda, const __bf16* W, int ldw,
                  float* C, int ldc, int M, int N, int K) {
    dim3 grid((N + 127) / 128, M / 128);
    gemm_bf16_wmma<<<grid, 256, 0, stream>>>(A, lda, W, ldw, C, ldc, M, N, K);
  };

  // in_proj: proj[BL, 2I]
  gemm(hs_bf, D_, inw_bf, D_, projf, 2 * I_, BL_, 2 * I_, D_);

  // causal depthwise conv + SiLU -> xc
  {
    size_t n = (size_t)BL_ * I_;
    conv_silu_kernel<<<(unsigned)((n + 255) / 256), 256, 0, stream>>>(
        projf, conv_w, conv_b, xcf, xc_bf);
  }

  // x_proj: xdbc[BL, 160]
  gemm(xc_bf, I_, xpw_bf, I_, xdbcf, XDBC_, BL_, XDBC_, I_);
  cvt(xdbcf, xdbc_bf, (size_t)BL_ * XDBC_);

  // dt_proj (uses first 128 cols of xdbc via lda=160) + softplus
  gemm(xdbc_bf, XDBC_, dtw_bf, DTR_, dtf, I_, BL_, I_, DTR_);
  {
    size_t n = (size_t)BL_ * I_;
    dt_softplus_kernel<<<(unsigned)((n + 255) / 256), 256, 0, stream>>>(dtf, dt_b);
  }

  // k / v projections
  gemm(hs_bf, D_, kw_bf, D_, kfb, I_, BL_, I_, D_);
  gemm(hs_bf, D_, vw_bf, D_, vfb, I_, BL_, I_, D_);

  // selective scan -> y_mamba
  scan_kernel<<<dim3(I_ / 256, B_), 256, 0, stream>>>(dtf, xcf, xdbcf, A_log, Dsk, ymf);

  // RoPE + pack q/k/v to [B,H,L,HD] bf16
  {
    size_t n = (size_t)B_ * H_ * L_ * HD_;
    rope_pack_kernel<<<(unsigned)((n + 255) / 256), 256, 0, stream>>>(
        xcf, kfb, vfb, qh, kh, vh);
  }

  // causal flash attention -> y_attn
  flash_attn_kernel<<<dim3(B_ * H_, L_ / 64), 128, 0, stream>>>(qh, kh, vh, yaf);

  // per-head RMSNorm + combine + SiLU(z) gate -> bf16
  norm_gate_kernel<<<BL_ * H_ / 8, 256, 0, stream>>>(
      yaf, ymf, attn_nw, mamba_nw, projf, ygate);

  // out_proj -> d_out
  gemm(ygate, I_, ow_bf, I_, out, D_, BL_, D_, I_);
}
